// MultiHeadAttention_16234976378957
// MI455X (gfx1250) — matrix operand order repaired and checked by reading, not compiled, not run on hardware
//
#include <hip/hip_runtime.h>
#include <hip/hip_bf16.h>

// Problem constants: B=2, S=2048, D=1024, H=16, DH=64
#define BB 2
#define SS 2048
#define DD 1024
#define HH 16
#define DHD 64
#define KK DD
#define MM (BB * SS)   // 4096 rows for the [BS, D] GEMMs
#define LDP 72         // padded LDS row pitch (bf16 elems) to spread banks

typedef __attribute__((ext_vector_type(16))) __bf16 v16bf;
typedef __attribute__((ext_vector_type(8)))  __bf16 v8bf;
typedef __attribute__((ext_vector_type(4)))  __bf16 v4bf;
typedef __attribute__((ext_vector_type(8)))  float  v8f;
typedef __attribute__((ext_vector_type(4)))  float  v4f;

static __device__ __forceinline__ v8f wmma_bf16(v16bf a, v16bf b, v8f c) {
  // D = A(16x32 bf16) * B(32x16 bf16) + C(16x16 f32)
  return __builtin_amdgcn_wmma_f32_16x16x32_bf16(
      /*neg_a=*/false, a, /*neg_b=*/false, b,
      /*c_mod=*/(short)0, c, /*reuse_a=*/false, /*reuse_b=*/false);
}

// A-fragment (M=16 x K=32) from row-major [rows][K] memory (K contiguous).
// Lanes 0-15 hold K = {0..7, 16..23}; lanes 16-31 hold K = {8..15, 24..31}.
static __device__ __forceinline__ v16bf load_afrag(const __bf16* base, int ld,
                                                   int row0, int k0, int lane) {
  const int r  = lane & 15;
  const int hh = lane >> 4;
  const __bf16* p = base + (size_t)(row0 + r) * (size_t)ld + k0 + 8 * hh;
  v8bf lo = *(const v8bf*)(p);
  v8bf hi = *(const v8bf*)(p + 16);
  v16bf out;
#pragma unroll
  for (int i = 0; i < 8; ++i) { out[i] = lo[i]; out[i + 8] = hi[i]; }
  return out;
}

// B-fragment (K=32 x N=16) computing A*W^T from row-major W[N][K] (K contiguous).
// Lanes 0-15 hold K = {0..7, 16..23}; lanes 16-31 hold K = {8..15, 24..31}.
static __device__ __forceinline__ v16bf load_bfrag(const __bf16* base, int ld,
                                                   int col0, int k0, int lane) {
  const int n  = lane & 15;
  const int hh = lane >> 4;
  const __bf16* p = base + (size_t)(col0 + n) * (size_t)ld + k0 + 8 * hh;
  v8bf lo = *(const v8bf*)(p);
  v8bf hi = *(const v8bf*)(p + 16);
  v16bf out;
#pragma unroll
  for (int i = 0; i < 8; ++i) { out[i] = lo[i]; out[i + 8] = hi[i]; }
  return out;
}

// ---------------- f32 -> bf16 conversion ----------------
__global__ void cvt_f32_bf16(const float* __restrict__ in, __bf16* __restrict__ out, int n4) {
  const int i = blockIdx.x * blockDim.x + threadIdx.x;
  if (i < n4) {
    const v4f x = ((const v4f*)in)[i];
    v4bf y;
#pragma unroll
    for (int j = 0; j < 4; ++j) y[j] = (__bf16)x[j];
    ((v4bf*)out)[i] = y;
  }
}

// ---------------- tiled WMMA GEMM: C = A[M,K] * W[N,K]^T + bias ----------------
// Wave tile 32x64 (2x4 WMMA accumulators); block = 8 waves (4M x 2N) -> 128x128.
// MODE 0: Q -> bf16 [BS,D] scaled by 1/sqrt(DH); 1: K -> bf16; 2: V -> bf16 [B,H,DH,S]; 3: O -> f32.
template <int MODE>
__global__ __launch_bounds__(256) void gemm_bt(const __bf16* __restrict__ A,
                                               const __bf16* __restrict__ W,
                                               const float* __restrict__ bias,
                                               void* __restrict__ Cout) {
  const int lane = threadIdx.x & 31;
  const int wave = threadIdx.x >> 5;
  const int wm = wave >> 1, wn = wave & 1;
  const int row0 = blockIdx.x * 128 + wm * 32;
  const int col0 = blockIdx.y * 128 + wn * 64;

  v8f acc[2][4] = {};
#pragma unroll 4
  for (int k0 = 0; k0 < KK; k0 += 32) {
    v16bf a0 = load_afrag(A, KK, row0,      k0, lane);
    v16bf a1 = load_afrag(A, KK, row0 + 16, k0, lane);
    v16bf b[4];
#pragma unroll
    for (int j = 0; j < 4; ++j) b[j] = load_bfrag(W, KK, col0 + 16 * j, k0, lane);
    if (k0 + 32 < KK) {
      __builtin_prefetch(A + (size_t)(row0 + lane) * KK + k0 + 32, 0, 1);
      __builtin_prefetch(W + (size_t)(col0 + lane) * KK + k0 + 32, 0, 1);
      __builtin_prefetch(W + (size_t)(col0 + 32 + lane) * KK + k0 + 32, 0, 1);
    }
#pragma unroll
    for (int j = 0; j < 4; ++j) {
      acc[0][j] = wmma_bf16(a0, b[j], acc[0][j]);
      acc[1][j] = wmma_bf16(a1, b[j], acc[1][j]);
    }
  }

  const int n16 = lane & 15, hh = lane >> 4;
#pragma unroll
  for (int im = 0; im < 2; ++im) {
#pragma unroll
    for (int jn = 0; jn < 4; ++jn) {
#pragma unroll
      for (int v = 0; v < 8; ++v) {
        const int m = row0 + im * 16 + hh * 8 + v;   // C layout: lanes16-31 = rows 8-15
        const int n = col0 + jn * 16 + n16;
        const float val = acc[im][jn][v] + bias[n];
        if (MODE == 0) {
          ((__bf16*)Cout)[(size_t)m * DD + n] = (__bf16)(val * 0.125f);  // 1/sqrt(64)
        } else if (MODE == 1) {
          ((__bf16*)Cout)[(size_t)m * DD + n] = (__bf16)val;
        } else if (MODE == 2) {
          const int b = m / SS, s = m % SS;
          const int h = n / DHD, dh = n % DHD;
          ((__bf16*)Cout)[(((size_t)b * HH + h) * DHD + dh) * SS + s] = (__bf16)val;
        } else {
          ((float*)Cout)[(size_t)m * DD + n] = val;
        }
      }
    }
  }
}

// ---------------- causal flash attention ----------------
// K/V tiles are staged once per 64-key block into LDS by the whole workgroup
// (4x less L2 traffic than per-wave fragment loads), then each wave builds its
// WMMA fragments with ds_load_b128. 8 score WMMAs + 8 P*V WMMAs per block.
template <bool MASKED>
static __device__ __forceinline__ void attn_block(
    int kb, int q0, int lane,
    const __bf16* __restrict__ ldsK,   // [64][LDP] keys x d
    const __bf16* __restrict__ ldsV,   // [64][LDP] dh x keys(tile-local)
    __bf16* __restrict__ ldsPw,        // [16][LDP] this wave's P tile
    const v16bf& qf0, const v16bf& qf1,
    v8f acc[4], float mrow[8], float lrow[8]) {
  const int n16 = lane & 15, hh = lane >> 4, mbase = hh * 8;

  // scores for 64 keys as four 16x16 f32 tiles (K fragments from LDS)
  v8f s[4];
#pragma unroll
  for (int j = 0; j < 4; ++j) {
    v8f t = {};
    t = wmma_bf16(qf0, load_bfrag(ldsK, LDP, 16 * j, 0,  lane), t);
    t = wmma_bf16(qf1, load_bfrag(ldsK, LDP, 16 * j, 32, lane), t);
    s[j] = t;
  }

  // (optional causal mask) + per-lane partial row max
  float pmax[8];
#pragma unroll
  for (int v = 0; v < 8; ++v) {
    const int qrow = q0 + mbase + v;
    float pm = -1e30f;
#pragma unroll
    for (int j = 0; j < 4; ++j) {
      float val = s[j][v];
      if (MASKED) val = (kb + 16 * j + n16 <= qrow) ? val : -1e30f;
      s[j][v] = val;
      pm = fmaxf(pm, val);
    }
    pmax[v] = pm;
  }
#pragma unroll
  for (int off = 1; off < 16; off <<= 1) {
#pragma unroll
    for (int v = 0; v < 8; ++v)
      pmax[v] = fmaxf(pmax[v], __shfl_xor(pmax[v], off, 32));
  }

  // online softmax update
  float alpha[8], rsum[8];
#pragma unroll
  for (int v = 0; v < 8; ++v) {
    const float nm = fmaxf(mrow[v], pmax[v]);
    alpha[v] = __expf(mrow[v] - nm);
    mrow[v] = nm;
    float rs = 0.0f;
#pragma unroll
    for (int j = 0; j < 4; ++j) {
      const float p = __expf(s[j][v] - nm);
      s[j][v] = p;
      rs += p;
    }
    rsum[v] = rs;
  }
#pragma unroll
  for (int off = 1; off < 16; off <<= 1) {
#pragma unroll
    for (int v = 0; v < 8; ++v)
      rsum[v] += __shfl_xor(rsum[v], off, 32);
  }
#pragma unroll
  for (int v = 0; v < 8; ++v) {
    lrow[v] = lrow[v] * alpha[v] + rsum[v];
#pragma unroll
    for (int i = 0; i < 4; ++i) acc[i][v] *= alpha[v];
  }

  // P (C-layout f32) -> LDS row-major bf16 (16x64) -> reload as two A-fragments
#pragma unroll
  for (int v = 0; v < 8; ++v) {
#pragma unroll
    for (int j = 0; j < 4; ++j)
      ldsPw[(mbase + v) * LDP + 16 * j + n16] = (__bf16)s[j][v];
  }
  asm volatile("s_wait_dscnt 0" ::: "memory");   // same-wave LDS RAW fence
  const v16bf pf0 = load_afrag(ldsPw, LDP, 0, 0,  lane);
  const v16bf pf1 = load_afrag(ldsPw, LDP, 0, 32, lane);

  // O += P(16x64) * V(64x64); V fragments from LDS (dh-major rows)
#pragma unroll
  for (int nb = 0; nb < 4; ++nb) {
    v8f t = acc[nb];
    t = wmma_bf16(pf0, load_bfrag(ldsV, LDP, nb * 16, 0,  lane), t);
    t = wmma_bf16(pf1, load_bfrag(ldsV, LDP, nb * 16, 32, lane), t);
    acc[nb] = t;
  }
}

// Grid: (S/64, H, B); block = 128 threads = 4 waves; each wave owns 16 query rows.
// Q,K: bf16 [B,S,D] per-head slices (row stride D); Vt: bf16 [B,H,DH,S].
__global__ __launch_bounds__(128) void flash_attn(const __bf16* __restrict__ Q,
                                                  const __bf16* __restrict__ Km,
                                                  const __bf16* __restrict__ Vt,
                                                  __bf16* __restrict__ O) {
  __shared__ __align__(16) __bf16 ldsK[64][LDP];       // K tile: 64 keys x 64 d
  __shared__ __align__(16) __bf16 ldsV[64][LDP];       // V tile: 64 dh  x 64 keys
  __shared__ __align__(16) __bf16 ldsP[4][16][LDP];    // per-wave P tiles
  const int lane  = threadIdx.x & 31;
  const int wave  = threadIdx.x >> 5;
  const int n16   = lane & 15;
  const int hh    = lane >> 4;
  const int mbase = hh * 8;
  const int qblk  = blockIdx.x;
  const int h     = blockIdx.y;
  const int b     = blockIdx.z;
  const int q0    = qblk * 64 + wave * 16;

  const __bf16* qp = Q  + (size_t)b * SS * DD + h * DHD;
  const __bf16* kp = Km + (size_t)b * SS * DD + h * DHD;
  const __bf16* vp = Vt + ((size_t)(b * HH + h)) * DHD * SS;
  __bf16* ldsPw = &ldsP[wave][0][0];

  const v16bf qf0 = load_afrag(qp, DD, q0, 0,  lane);   // d = 0..31
  const v16bf qf1 = load_afrag(qp, DD, q0, 32, lane);   // d = 32..63

  v8f acc[4] = {};                      // O tile: 16 rows x 64 dh (4 N-blocks)
  float mrow[8], lrow[8];
#pragma unroll
  for (int v = 0; v < 8; ++v) { mrow[v] = -1e30f; lrow[v] = 0.0f; }

  const int kend = qblk * 64 + 64;      // same trip count for all 4 waves

  for (int kb = 0; kb < kend; kb += 64) {
    // cooperative staging: 64 rows x 32 bf16 (four 8-elem chunks, stride 16) per tile, 2x b128 per thread per tile
#pragma unroll
    for (int t = 0; t < 2; ++t) {
      const int idx = threadIdx.x + t * 128;   // 0..255 -> row=idx/4, chunk=idx%4
      const int row = idx >> 2, c = (idx & 3) * 16;
      *(v8bf*)&ldsK[row][c] = *(const v8bf*)(kp + (size_t)(kb + row) * DD + c);
      *(v8bf*)&ldsV[row][c] = *(const v8bf*)(vp + (size_t)row * SS + kb + c);
    }
    __syncthreads();

    if (kb + 63 <= q0)   // entire 64-key block below this wave's causal frontier
      attn_block<false>(kb, q0, lane, &ldsK[0][0], &ldsV[0][0], ldsPw,
                        qf0, qf1, acc, mrow, lrow);
    else                 // boundary (or fully-masked) block: elementwise mask
      attn_block<true>(kb, q0, lane, &ldsK[0][0], &ldsV[0][0], ldsPw,
                       qf0, qf1, acc, mrow, lrow);
    __syncthreads();     // tiles consumed before next staging round
  }

  // epilogue: one reciprocal per row, then scale + store bf16 [B,S,D]
  float linv[8];
#pragma unroll
  for (int v = 0; v < 8; ++v) linv[v] = __builtin_amdgcn_rcpf(lrow[v]);
#pragma unroll
  for (int nb = 0; nb < 4; ++nb) {
#pragma unroll
    for (int v = 0; v < 8; ++v) {
      const int row = q0 + mbase + v;
      const int col = h * DHD + nb * 16 + n16;
      O[((size_t)b * SS + row) * DD + col] = (__bf16)(acc[nb][v] * linv[v]);
    }
  }
}

extern "C" void kernel_launch(void* const* d_in, const int* in_sizes, int n_in,
                              void* d_out, int out_size, void* d_ws, size_t ws_size,
                              hipStream_t stream) {
  (void)in_sizes; (void)n_in; (void)out_size; (void)ws_size;
  const float* x  = (const float*)d_in[0];
  // d_in[1] = causal mask (tril) -- computed analytically in-kernel
  const float* Wq = (const float*)d_in[2];
  const float* bq = (const float*)d_in[3];
  const float* Wk = (const float*)d_in[4];
  const float* bk = (const float*)d_in[5];
  const float* Wv = (const float*)d_in[6];
  const float* bv = (const float*)d_in[7];
  const float* Wo = (const float*)d_in[8];
  const float* bo = (const float*)d_in[9];

  char* ws = (char*)d_ws;
  const size_t szX = (size_t)MM * DD * sizeof(__bf16);   // 8 MB
  const size_t szW = (size_t)DD * DD * sizeof(__bf16);   // 2 MB
  __bf16* xb  = (__bf16*)(ws);
  __bf16* wqb = (__bf16*)(ws + szX);
  __bf16* wkb = (__bf16*)(ws + szX + 1 * szW);
  __bf16* wvb = (__bf16*)(ws + szX + 2 * szW);
  __bf16* wob = (__bf16*)(ws + szX + 3 * szW);
  __bf16* qb  = (__bf16*)(ws + 1 * szX + 4 * szW);
  __bf16* kb  = (__bf16*)(ws + 2 * szX + 4 * szW);
  __bf16* vtb = (__bf16*)(ws + 3 * szX + 4 * szW);
  __bf16* ab  = (__bf16*)(ws + 4 * szX + 4 * szW);     // total 48 MB

  const int nx4 = MM * DD / 4;
  const int nw4 = DD * DD / 4;
  cvt_f32_bf16<<<(nx4 + 255) / 256, 256, 0, stream>>>(x,  xb,  nx4);
  cvt_f32_bf16<<<(nw4 + 255) / 256, 256, 0, stream>>>(Wq, wqb, nw4);
  cvt_f32_bf16<<<(nw4 + 255) / 256, 256, 0, stream>>>(Wk, wkb, nw4);
  cvt_f32_bf16<<<(nw4 + 255) / 256, 256, 0, stream>>>(Wv, wvb, nw4);
  cvt_f32_bf16<<<(nw4 + 255) / 256, 256, 0, stream>>>(Wo, wob, nw4);

  const dim3 ggrid(MM / 128, DD / 128);  // (32, 8)
  gemm_bt<0><<<ggrid, 256, 0, stream>>>(xb, wqb, bq, (void*)qb);
  gemm_bt<1><<<ggrid, 256, 0, stream>>>(xb, wkb, bk, (void*)kb);
  gemm_bt<2><<<ggrid, 256, 0, stream>>>(xb, wvb, bv, (void*)vtb);

  const dim3 fgrid(SS / 64, HH, BB);     // (32, 16, 2)
  flash_attn<<<fgrid, 128, 0, stream>>>(qb, kb, vtb, ab);

  gemm_bt<3><<<ggrid, 256, 0, stream>>>(ab, wob, bo, d_out);
}
